// MixAdapter_19920058319071
// MI455X (gfx1250) — matrix operand order repaired and checked by reading, not compiled, not run on hardware
//
#include <hip/hip_runtime.h>
#include <hip/hip_bf16.h>

// ---------------------------------------------------------------------------
// MixAdapter on MI455X (gfx1250, wave32, WMMA + TDM).
// Roofline: ~27 GFLOP, ~210MB HBM traffic -> bf16 WMMA (16x16x32, f32 acc)
// puts GEMM time ~= memory time (~10us class). fp32 WMMA would be ~8x slower
// per issue and compute-bound. Residual + LN kept in fp32.
// A-strips are staged to LDS via the Tensor Data Mover (tensor_load_to_lds,
// TENSORcnt), B streams from L2 (weights ~1.7MB bf16, L2-resident).
// ---------------------------------------------------------------------------

typedef __bf16 bf16_t;
typedef __attribute__((ext_vector_type(16))) __bf16 v16bf;
typedef __attribute__((ext_vector_type(8)))  __bf16 v8bf;
typedef __attribute__((ext_vector_type(4)))  __bf16 v4bf;
typedef __attribute__((ext_vector_type(8)))  float  v8f;
typedef __attribute__((ext_vector_type(4)))  unsigned int u32x4;
typedef __attribute__((ext_vector_type(8)))  int i32x8;
typedef __attribute__((ext_vector_type(4)))  int i32x4;

#define D_EMB   1024
#define BOT     400
#define BOTP    416          // BOT padded to multiple of 32 (WMMA K step)
#define NADAPT  25
#define BATCH   8
#define SEQ     2048
#define MTOT    (BATCH * SEQ)   // 16384 rows
#define LNEPS   1e-5f
#define CHUNKS  64              // stats partial-reduction chunks per batch

#ifndef __has_builtin
#define __has_builtin(x) 0
#endif
#if __has_builtin(__builtin_amdgcn_tensor_load_to_lds)
#define USE_TDM 1
#else
#define USE_TDM 0
#endif

// ---------------- adapter superposition (alpha-weighted merge) --------------

__global__ void merge_down_k(const float* __restrict__ Wall,
                             const float* __restrict__ alphas,
                             bf16_t* __restrict__ Wd) {
  int idx = blockIdx.x * blockDim.x + threadIdx.x;   // over BOTP * D
  if (idx >= BOTP * D_EMB) return;
  int r = idx / D_EMB, c = idx % D_EMB;
  float s = 0.f;
  if (r < BOT) {
    const float* p = Wall + (size_t)r * D_EMB + c;
    for (int n = 0; n < NADAPT; ++n)
      s += alphas[n] * p[(size_t)n * BOT * D_EMB];
  }
  Wd[idx] = (bf16_t)s;   // padded rows [400,416) -> 0
}

__global__ void merge_up_k(const float* __restrict__ Wall,
                           const float* __restrict__ alphas,
                           bf16_t* __restrict__ Wu) {
  int idx = blockIdx.x * blockDim.x + threadIdx.x;   // over D * BOTP
  if (idx >= D_EMB * BOTP) return;
  int d = idx / BOTP, o = idx % BOTP;
  float s = 0.f;
  if (o < BOT) {
    const float* p = Wall + (size_t)d * BOT + o;
    for (int n = 0; n < NADAPT; ++n)
      s += alphas[n] * p[(size_t)n * D_EMB * BOT];
  }
  Wu[idx] = (bf16_t)s;   // padded cols [400,416) -> 0
}

__global__ void merge_ln_k(const float* __restrict__ Wlnall,
                           const float* __restrict__ blnall,
                           const float* __restrict__ alphas,
                           float* __restrict__ wln, float* __restrict__ bln) {
  int d = blockIdx.x * blockDim.x + threadIdx.x;
  if (d >= D_EMB) return;
  float sw = 0.f, sb = 0.f;
  for (int n = 0; n < NADAPT; ++n) {
    float a = alphas[n];
    sw += a * Wlnall[n * D_EMB + d];
    sb += a * blnall[n * D_EMB + d];
  }
  wln[d] = sw; bln[d] = sb;
}

// ---------------- per-batch layernorm stats (deterministic 2-stage) ---------

__global__ void stats_partial_k(const float* __restrict__ x,
                                float* __restrict__ part) {
  __shared__ float s1[256], s2[256];
  const int b = blockIdx.y, chunk = blockIdx.x;
  const size_t per = (size_t)SEQ * D_EMB;            // 2M per batch
  const size_t cs  = per / CHUNKS;
  const float* p = x + (size_t)b * per + (size_t)chunk * cs;
  float s = 0.f, ss = 0.f;
  for (size_t i = threadIdx.x; i < cs; i += 256) {
    float v = p[i]; s += v; ss += v * v;
  }
  s1[threadIdx.x] = s; s2[threadIdx.x] = ss;
  __syncthreads();
  for (int o = 128; o > 0; o >>= 1) {
    if (threadIdx.x < o) {
      s1[threadIdx.x] += s1[threadIdx.x + o];
      s2[threadIdx.x] += s2[threadIdx.x + o];
    }
    __syncthreads();
  }
  if (threadIdx.x == 0) {
    part[(b * CHUNKS + chunk) * 2 + 0] = s1[0];
    part[(b * CHUNKS + chunk) * 2 + 1] = s2[0];
  }
}

__global__ void stats_final_k(const float* __restrict__ part,
                              float* __restrict__ stats) {
  __shared__ float s1[64], s2[64];
  const int b = blockIdx.x;
  float s = 0.f, ss = 0.f;
  for (int i = threadIdx.x; i < CHUNKS; i += 64) {
    s  += part[(b * CHUNKS + i) * 2 + 0];
    ss += part[(b * CHUNKS + i) * 2 + 1];
  }
  s1[threadIdx.x] = s; s2[threadIdx.x] = ss;
  __syncthreads();
  for (int o = 32; o > 0; o >>= 1) {
    if (threadIdx.x < o) {
      s1[threadIdx.x] += s1[threadIdx.x + o];
      s2[threadIdx.x] += s2[threadIdx.x + o];
    }
    __syncthreads();
  }
  if (threadIdx.x == 0) {
    const float inv = 1.0f / ((float)SEQ * (float)D_EMB);
    float mean = s1[0] * inv;
    float var  = s2[0] * inv - mean * mean;
    stats[b * 2 + 0] = mean;
    stats[b * 2 + 1] = rsqrtf(var + LNEPS);
  }
}

// ---------------- normalize + scale/shift -> bf16 ---------------------------

__global__ void normalize_k(const float4* __restrict__ x,
                            const float* __restrict__ stats,
                            const float4* __restrict__ wln,
                            const float4* __restrict__ bln,
                            v4bf* __restrict__ xn) {
  unsigned i = blockIdx.x * blockDim.x + threadIdx.x;  // over MTOT*D/4
  int dq = i & (D_EMB / 4 - 1);
  int b  = i >> 19;                                    // SEQ*D/4 = 2^19
  float mean = stats[2 * b + 0];
  float rstd = stats[2 * b + 1];
  float4 v = x[i];
  float4 w = wln[dq];
  float4 bb = bln[dq];
  v4bf o;
  o[0] = (bf16_t)((v.x - mean) * rstd * w.x + bb.x);
  o[1] = (bf16_t)((v.y - mean) * rstd * w.y + bb.y);
  o[2] = (bf16_t)((v.z - mean) * rstd * w.z + bb.z);
  o[3] = (bf16_t)((v.w - mean) * rstd * w.w + bb.w);
  xn[i] = o;
}

// ---------------- bf16 WMMA GEMM ( D = A * B^T_rowmajor ) -------------------
// A: [MTOT, KDIM] bf16 row-major (strip staged in LDS via TDM)
// Bm: [NTILES*16, KDIM] bf16 row-major  (B[k,n] = Bm[n*KDIM + k])
// RELU_BF16 ? out = relu(acc) as bf16 : out = resid + acc as f32
// Block = 256 thr = 8 waves; one 16-row M strip per block; waves stride over N.

template <int KDIM, int NTILES, bool RELU_BF16>
__global__ void __launch_bounds__(256)
wmma_gemm_k(const bf16_t* __restrict__ A,
            const bf16_t* __restrict__ Bm,
            const float* __restrict__ resid,
            bf16_t* __restrict__ outb,
            float* __restrict__ outf) {
  __shared__ alignas(16) bf16_t lds[16 * KDIM];

  const int m0 = blockIdx.x * 16;

#if USE_TDM
  // Stage the 16 x KDIM strip (16*KDIM/2 DWORDs, contiguous) with the Tensor
  // Data Mover: one wave issues a 1-D D# (data_size=4B, tile_dim0 = NDW),
  // waits TENSORcnt==0, then the workgroup barrier publishes the LDS data.
  if (threadIdx.x < 32) {
    constexpr unsigned NDW = (unsigned)(16 * KDIM / 2);   // DWORDs to move
    unsigned long long ga =
        (unsigned long long)(const void*)(A + (size_t)m0 * KDIM);
    unsigned ldsoff = (unsigned)(unsigned long long)(void*)&lds[0]; // addr[31:0]
    u32x4 g0;
    g0[0] = 1u;                                   // count=1 valid descriptor
    g0[1] = ldsoff;                               // lds_addr
    g0[2] = (unsigned)ga;                         // global_addr[31:0]
    g0[3] = (unsigned)((ga >> 32) & 0x01FFFFFFu)  // global_addr[56:32]
            | (2u << 30);                         // type=2 ("image")
    i32x8 g1;
    g1[0] = 0x00020000;                           // data_size=2 (4B), mask=0
    g1[1] = (int)(NDW << 16);                     // tensor_dim0[15:0]
    g1[2] = (int)(1u << 16);                      // tensor_dim0[31:16]=0, tensor_dim1=1
    g1[3] = (int)(NDW << 16);                     // tile_dim0 = NDW
    g1[4] = 0;                                    // tile_dim1/2 unused
    g1[5] = (int)NDW;                             // tensor_dim0_stride[31:0]
    g1[6] = 0;
    g1[7] = 0;
    i32x4 z4 = {0, 0, 0, 0};                      // <=2D: groups 2/3 unused
#if defined(__clang_major__) && __clang_major__ >= 23
    i32x8 z8 = {0, 0, 0, 0, 0, 0, 0, 0};
    __builtin_amdgcn_tensor_load_to_lds(g0, g1, z4, z4, z8, 0);
#else
    __builtin_amdgcn_tensor_load_to_lds(g0, g1, z4, z4, 0);
#endif
    __builtin_amdgcn_s_wait_tensorcnt(0);
  }
#else
  {
    const v8bf* src = (const v8bf*)(A + (size_t)m0 * KDIM);
    v8bf* dst = (v8bf*)lds;
    const int total = 16 * KDIM / 8;
    for (int i = threadIdx.x; i < total; i += 256) dst[i] = src[i];
  }
#endif
  __syncthreads();

  const int lane = threadIdx.x & 31;
  const int wave = threadIdx.x >> 5;
  const int arow = lane & 15;           // A-frag: row within tile
  const int akof = (lane >> 4) * 8;     // A-frag: K sub-offset {0,8}
  const int ncol = lane & 15;           // B/C: column within tile
  const int bkof = (lane >> 4) * 8;     // B-frag: K sub-offset {0,8}
  const int rbase = (lane >> 4) * 8;    // C/D: row base {0,8}

  const bf16_t* aptr = lds + arow * KDIM + akof;

  for (int nt = wave; nt < NTILES; nt += 8) {
    v8f acc = {};
    const bf16_t* bptr = Bm + (size_t)(nt * 16 + ncol) * KDIM + bkof;
    for (int k0 = 0; k0 < KDIM; k0 += 32) {
      __builtin_prefetch(bptr + k0 + 128, 0, 1);      // global_prefetch_b8
      v8bf alo = *(const v8bf*)(aptr + k0);           // K {0..7}+akof
      v8bf ahi = *(const v8bf*)(aptr + k0 + 16);      // K {16..23}+akof
      v16bf a = __builtin_shufflevector(alo, ahi,
                  0, 1, 2, 3, 4, 5, 6, 7, 8, 9, 10, 11, 12, 13, 14, 15);
      v8bf blo = *(const v8bf*)(bptr + k0);           // K {0..7}+bkof
      v8bf bhi = *(const v8bf*)(bptr + k0 + 16);      // K {16..23}+bkof
      v16bf b = __builtin_shufflevector(blo, bhi,
                  0, 1, 2, 3, 4, 5, 6, 7, 8, 9, 10, 11, 12, 13, 14, 15);
      acc = __builtin_amdgcn_wmma_f32_16x16x32_bf16(
          /*neg_a=*/false, a, /*neg_b=*/false, b,
          /*c_mod=*/(short)0, acc, /*reuse_a=*/false, /*reuse_b=*/false);
    }
    const int ostride = NTILES * 16;
    if (RELU_BF16) {
      for (int r = 0; r < 8; ++r) {
        float v = acc[r];
        v = v > 0.f ? v : 0.f;
        outb[(size_t)(m0 + rbase + r) * ostride + nt * 16 + ncol] = (bf16_t)v;
      }
    } else {
      for (int r = 0; r < 8; ++r) {
        size_t o = (size_t)(m0 + rbase + r) * ostride + nt * 16 + ncol;
        outf[o] = resid[o] + acc[r];
      }
    }
  }
}

// ---------------------------------------------------------------------------

extern "C" void kernel_launch(void* const* d_in, const int* in_sizes, int n_in,
                              void* d_out, int out_size, void* d_ws, size_t ws_size,
                              hipStream_t stream) {
  const float* x      = (const float*)d_in[0];
  const float* alphas = (const float*)d_in[1];
  const float* Wdall  = (const float*)d_in[2];
  const float* Wuall  = (const float*)d_in[3];
  const float* Wlnall = (const float*)d_in[6];
  const float* blnall = (const float*)d_in[7];
  float* out = (float*)d_out;

  // workspace layout (bytes)
  char* w = (char*)d_ws;
  bf16_t* Xn    = (bf16_t*)(w);                         // 16384*1024*2 = 33,554,432
  bf16_t* H     = (bf16_t*)(w + 33554432);              // 16384*416*2  = 13,631,488
  bf16_t* Wd    = (bf16_t*)(w + 47185920);              // 416*1024*2   =    851,968
  bf16_t* Wu    = (bf16_t*)(w + 48037888);              // 1024*416*2   =    851,968
  float*  wln   = (float*) (w + 48889856);              // 4096
  float*  bln   = (float*) (w + 48893952);              // 4096
  float*  part  = (float*) (w + 48898048);              // 8*64*2*4 = 4096
  float*  stats = (float*) (w + 48902144);              // 64

  // 1) merge adapters
  merge_down_k<<<(BOTP * D_EMB + 255) / 256, 256, 0, stream>>>(Wdall, alphas, Wd);
  merge_up_k  <<<(D_EMB * BOTP + 255) / 256, 256, 0, stream>>>(Wuall, alphas, Wu);
  merge_ln_k  <<<(D_EMB + 255) / 256,        256, 0, stream>>>(Wlnall, blnall, alphas, wln, bln);

  // 2) per-batch LN stats over (S, D)
  stats_partial_k<<<dim3(CHUNKS, BATCH), 256, 0, stream>>>(x, part);
  stats_final_k  <<<BATCH, 64, 0, stream>>>(part, stats);

  // 3) normalize -> bf16
  normalize_k<<<(MTOT * D_EMB / 4) / 256, 256, 0, stream>>>(
      (const float4*)x, stats, (const float4*)wln, (const float4*)bln, (v4bf*)Xn);

  // 4) H = relu(Xn @ Wd^T)   [16384, 416] bf16 (padding cols auto-zero)
  wmma_gemm_k<D_EMB, BOTP / 16, true><<<MTOT / 16, 256, 0, stream>>>(
      Xn, Wd, nullptr, H, nullptr);

  // 5) out = x + H @ Wu^T    [16384, 1024] f32
  wmma_gemm_k<BOTP, D_EMB / 16, false><<<MTOT / 16, 256, 0, stream>>>(
      H, Wu, x, nullptr, out);
}
